// pka_GNN_18219251269758
// MI455X (gfx1250) — matrix operand order repaired and checked by reading, not compiled, not run on hardware
//
#include <hip/hip_runtime.h>
#include <hip/hip_bf16.h>
#include <stdint.h>

#define N_NODES 100000
#define N_EDGES 200000
#define HID 128

typedef __attribute__((ext_vector_type(16))) __bf16 v16bf;
typedef __attribute__((ext_vector_type(8)))  __bf16 v8bf;
typedef __attribute__((ext_vector_type(8)))  float  v8f;

union FragBF { v16bf v; v8bf h[2]; };

// ---------------- weight convert f32 -> bf16 with K padding ----------------
__global__ void k_cvt_w(const float* __restrict__ w, __bf16* __restrict__ o,
                        int nout, int k, int kpad) {
  int i = blockIdx.x * 256 + threadIdx.x;
  int total = nout * kpad;
  if (i >= total) return;
  int r = i / kpad, c = i - r * kpad;
  float v = (c < k) ? w[(size_t)r * k + c] : 0.f;
  o[i] = (__bf16)v;
}

__global__ void k_zero_f(float* __restrict__ p, int n) {
  int i = blockIdx.x * 256 + threadIdx.x;
  if (i < n) p[i] = 0.f;
}

// ---------------- WMMA GEMM: Y = epilogue(A @ W^T + b) ---------------------
// A is staged into LDS as bf16 with a fused gather per AMODE:
//   AMODE 0: A[i]   = X[i]                         (K=128)
//   AMODE 1: A[e]   = concat(X[src[e]], EA[e], 0)  (K=160, real 144)
//   AMODE 2: A[e]   = ACC[src[e]] - He[rev[e]]     (K=128)
//   AMODE 3: A[i]   = concat(X[i], ACC[i])         (K=256)
// EMODE 0: OUTF = relu(y+b)            (f32)
// EMODE 1: H0OUT = y+b ; HOUT = relu(y+b)          (bf16, bf16)
// EMODE 2: HOUT = relu(H0IN + y + b)               (bf16)
// Block: 256 threads = 8 waves; wave w owns output cols [16w,16w+16);
// block owns 32 output rows (two 16x16 tiles per wave, B reused).
template<int AMODE, int EMODE, int KT>
__global__ __launch_bounds__(256) void k_gemm(
    const float* __restrict__ X, const float* __restrict__ EA,
    const float* __restrict__ ACC, const __bf16* __restrict__ HREV,
    const int* __restrict__ SRC, const int* __restrict__ REV,
    const __bf16* __restrict__ W, const float* __restrict__ BIAS,
    const __bf16* __restrict__ H0IN,
    float* __restrict__ OUTF, __bf16* __restrict__ H0OUT, __bf16* __restrict__ HOUT,
    int M)
{
  __shared__ __bf16 As[32 * KT];
  const int tid = threadIdx.x;
  const int m0  = blockIdx.x * 32;

  // ---- stage A tile (32 x KT) into LDS as bf16, gather fused ----
  for (int i = tid; i < 32 * KT; i += 256) {
    const int r = i / KT, c = i - r * KT;
    const int row = m0 + r;
    float v = 0.f;
    if (row < M) {
      if (AMODE == 0) {
        v = X[(size_t)row * HID + c];
      } else if (AMODE == 1) {
        const int s = SRC[row];
        if (c < HID)           v = X[(size_t)s * HID + c];
        else if (c < HID + 16) v = EA[(size_t)row * 16 + (c - HID)];
      } else if (AMODE == 2) {
        const int s  = SRC[row];
        const int rv = REV[row];
        v = ACC[(size_t)s * HID + c];
        if (rv >= 0) v -= (float)HREV[(size_t)rv * HID + c];
      } else {
        v = (c < HID) ? X[(size_t)row * HID + c]
                      : ACC[(size_t)row * HID + (c - HID)];
      }
    }
    As[i] = (__bf16)v;
  }
  __syncthreads();

  const int lane = tid & 31;
  const int wave = tid >> 5;
  const int nloc = lane & 15;   // output col within tile / A row (M) index
  const int half = lane >> 4;   // K half-selector per WMMA layout
  const int n    = wave * 16 + nloc;   // global output col (0..127)

  v8f acc0 = {}; v8f acc1 = {};
  constexpr int KSTEPS = KT / 32;
#pragma unroll
  for (int kc = 0; kc < KSTEPS; ++kc) {
    FragBF b, a0, a1;
    // B (32x16, K x N): half0 -> K {0..7,16..23}, half1 -> K {8..15,24..31}
    const __bf16* bp = W + (size_t)n * KT + kc * 32 + half * 8;
    b.h[0] = *(const v8bf*)(bp);
    b.h[1] = *(const v8bf*)(bp + 16);
    // A (16x32): half0 -> K {0..7,16..23}, half1 -> K {8..15,24..31}
    const __bf16* ap0 = &As[(size_t)nloc * KT + kc * 32 + half * 8];
    a0.h[0] = *(const v8bf*)(ap0);
    a0.h[1] = *(const v8bf*)(ap0 + 16);
    const __bf16* ap1 = &As[(size_t)(16 + nloc) * KT + kc * 32 + half * 8];
    a1.h[0] = *(const v8bf*)(ap1);
    a1.h[1] = *(const v8bf*)(ap1 + 16);
    acc0 = __builtin_amdgcn_wmma_f32_16x16x32_bf16(false, a0.v, false, b.v,
                                                   (short)0, acc0, false, false);
    acc1 = __builtin_amdgcn_wmma_f32_16x16x32_bf16(false, a1.v, false, b.v,
                                                   (short)0, acc1, false, false);
  }

  const float bv = BIAS[n];
#pragma unroll
  for (int r = 0; r < 8; ++r) {
    const int rowl = r + 8 * half;      // C/D layout: M = r + 8*half
    const int row0 = m0 + rowl;
    const int row1 = m0 + 16 + rowl;
    float y0 = acc0[r] + bv;
    float y1 = acc1[r] + bv;
    if (EMODE == 0) {
      if (row0 < M) OUTF[(size_t)row0 * HID + n] = fmaxf(y0, 0.f);
      if (row1 < M) OUTF[(size_t)row1 * HID + n] = fmaxf(y1, 0.f);
    } else if (EMODE == 1) {
      if (row0 < M) { H0OUT[(size_t)row0*HID+n] = (__bf16)y0;
                      HOUT [(size_t)row0*HID+n] = (__bf16)fmaxf(y0, 0.f); }
      if (row1 < M) { H0OUT[(size_t)row1*HID+n] = (__bf16)y1;
                      HOUT [(size_t)row1*HID+n] = (__bf16)fmaxf(y1, 0.f); }
    } else {
      if (row0 < M) { y0 += (float)H0IN[(size_t)row0*HID+n];
                      HOUT[(size_t)row0*HID+n] = (__bf16)fmaxf(y0, 0.f); }
      if (row1 < M) { y1 += (float)H0IN[(size_t)row1*HID+n];
                      HOUT[(size_t)row1*HID+n] = (__bf16)fmaxf(y1, 0.f); }
    }
  }
}

// ---------------- scatter: acc[dst[e]] += He[gidx?gidx[e]:e] ---------------
__global__ void k_scatter(const __bf16* __restrict__ He, const int* __restrict__ gidx,
                          const int* __restrict__ dst, float* __restrict__ acc) {
  long i = (long)blockIdx.x * 256 + threadIdx.x;
  const long total = (long)N_EDGES * HID;
  if (i >= total) return;
  const int e = (int)(i >> 7);
  const int c = (int)(i & 127);
  const int r = gidx ? gidx[e] : e;
  atomicAdd(acc + (size_t)dst[e] * HID + c, (float)He[(size_t)r * HID + c]);
}

// ---------------- iso select: Msel = (rowsum(Magg)==0) ? x : Magg ----------
__global__ void k_msel(float* __restrict__ magg, const float* __restrict__ xb) {
  const int gw   = (blockIdx.x * 256 + threadIdx.x) >> 5;  // one wave per row
  const int lane = threadIdx.x & 31;
  if (gw >= N_NODES) return;
  const size_t base = (size_t)gw * HID + lane;
  float v0 = magg[base], v1 = magg[base + 32], v2 = magg[base + 64], v3 = magg[base + 96];
  float s = v0 + v1 + v2 + v3;
  for (int off = 16; off > 0; off >>= 1) s += __shfl_xor(s, off, 32);
  if (s == 0.f) {
    magg[base]      = xb[base];
    magg[base + 32] = xb[base + 32];
    magg[base + 64] = xb[base + 64];
    magg[base + 96] = xb[base + 96];
  }
}

// ---------------- classifier head c2 (2 outputs) ---------------------------
__global__ void k_logits(const float* __restrict__ t, const float* __restrict__ w2,
                         const float* __restrict__ b2, float* __restrict__ logits) {
  const int i = blockIdx.x * 256 + threadIdx.x;
  if (i >= N_NODES) return;
  float a0 = b2[0], a1 = b2[1];
#pragma unroll 4
  for (int k = 0; k < HID; ++k) {
    const float tv = t[(size_t)i * HID + k];
    a0 += tv * w2[k];
    a1 += tv * w2[HID + k];
  }
  logits[2 * i] = a0; logits[2 * i + 1] = a1;
}

// ---------------- weighted NLL partial sum ---------------------------------
__global__ void k_cla(const float* __restrict__ logits, const int* __restrict__ idxs,
                      int kiter, float ratio, float* __restrict__ accum) {
  __shared__ float red[256];
  const int idx = idxs[kiter];
  float s = 0.f;
  for (int i = blockIdx.x * 256 + threadIdx.x; i < N_NODES; i += gridDim.x * 256) {
    const float l0 = logits[2 * i], l1 = logits[2 * i + 1];
    const float m   = fmaxf(l0, l1);
    const float lse = m + logf(expf(l0 - m) + expf(l1 - m));
    s += (i == idx) ? ratio * (lse - l1) : (lse - l0);
  }
  red[threadIdx.x] = s; __syncthreads();
  for (int o = 128; o > 0; o >>= 1) {
    if (threadIdx.x < o) red[threadIdx.x] += red[threadIdx.x + o];
    __syncthreads();
  }
  if (threadIdx.x == 0) atomicAdd(accum, red[0]);
}

// ---------------- single-row regressor + gate ------------------------------
__global__ void k_rowhead(const float* __restrict__ hcur, const int* __restrict__ idxs,
                          int kiter, const float* __restrict__ pka,
                          const float* __restrict__ r1w, const float* __restrict__ r1b,
                          const float* __restrict__ r2w, const float* __restrict__ r2b,
                          const float* __restrict__ gw,  const float* __restrict__ gb,
                          float* __restrict__ hrow, float* __restrict__ regacc) {
  __shared__ float h[HID];
  __shared__ float red[HID];
  const int idx = idxs[kiter];
  const int tid = threadIdx.x;               // 128 threads
  h[tid] = hcur[(size_t)idx * HID + tid];
  __syncthreads();
  float a = r1b[tid];
  float g = gb[tid];
  for (int k = 0; k < HID; ++k) {
    a += r1w[(size_t)tid * HID + k] * h[k];
    g += gw [(size_t)tid * HID + k] * h[k];
  }
  const float tv = fmaxf(a, 0.f);
  const float gv = tanhf(g);
  hrow[tid] = h[tid] * gv + h[tid];
  red[tid] = tv * r2w[tid];
  __syncthreads();
  for (int o = 64; o > 0; o >>= 1) {
    if (tid < o) red[tid] += red[tid + o];
    __syncthreads();
  }
  if (tid == 0) {
    const float pred = red[0] + r2b[0];
    const float d  = (pred - 7.0f) / 3.0f - (pka[idx] - 7.0f) / 3.0f;
    const float ad = fabsf(d);
    const float L  = (ad < 1.0f) ? 0.5f * d * d : (ad - 0.5f);
    atomicAdd(regacc, L);
  }
}

// ---------------- h_upd = h_static with row idx replaced -------------------
__global__ void k_copypatch(const float* __restrict__ hst, const float* __restrict__ hrow,
                            const int* __restrict__ idxs, int kiter,
                            float* __restrict__ hupd) {
  const int i = blockIdx.x * 256 + threadIdx.x;
  if (i >= N_NODES * HID) return;
  const int row = i >> 7;
  hupd[i] = (row == idxs[kiter]) ? hrow[i & 127] : hst[i];
}

// ---------------- final pka head r2 ----------------------------------------
__global__ void k_pka(const float* __restrict__ t, const float* __restrict__ r2w,
                      const float* __restrict__ r2b, float* __restrict__ out) {
  const int i = blockIdx.x * 256 + threadIdx.x;
  if (i >= N_NODES) return;
  float a = r2b[0];
#pragma unroll 4
  for (int k = 0; k < HID; ++k) a += t[(size_t)i * HID + k] * r2w[k];
  out[i] = a;
}

// ---------------- positive-label discovery + sort-by-value -----------------
__global__ void k_findpos(const float* __restrict__ pka, int* __restrict__ cnt,
                          int* __restrict__ poss, float* __restrict__ vals) {
  const int i = blockIdx.x * 256 + threadIdx.x;
  if (i >= N_NODES) return;
  const float v = pka[i];
  if (v > 0.f) {
    const int p = atomicAdd(cnt, 1);
    if (p < 8) { poss[p] = i; vals[p] = v; }
  }
}

__global__ void k_sort3(const int* __restrict__ cnt, const int* __restrict__ poss,
                        const float* __restrict__ vals, int* __restrict__ out) {
  if (threadIdx.x != 0 || blockIdx.x != 0) return;
  int n = *cnt; if (n > 8) n = 8;
  int p[8]; float v[8];
  for (int i = 0; i < n; ++i) { p[i] = poss[i]; v[i] = vals[i]; }
  for (int a = 0; a < n; ++a)
    for (int b = a + 1; b < n; ++b)
      if (v[b] < v[a] || (v[b] == v[a] && p[b] < p[a])) {
        float tv = v[a]; v[a] = v[b]; v[b] = tv;
        int   tp = p[a]; p[a] = p[b]; p[b] = tp;
      }
  for (int k = 0; k < 3; ++k) out[k] = (k < n) ? p[k] : 0;
}

__global__ void k_final(const float* __restrict__ cla, const float* __restrict__ reg,
                        float* __restrict__ out3) {
  if (threadIdx.x != 0 || blockIdx.x != 0) return;
  const float lc = *cla / (3.0f * (float)N_NODES);
  const float lr = *reg / 3.0f;
  out3[0] = lc + lr;  // total
  out3[1] = lc;       // loss_cla
  out3[2] = lr;       // loss_reg
}

// ===========================================================================
static inline void* ws_carve(char*& p, size_t bytes) {
  void* r = (void*)p;
  p += (bytes + 255) & ~(size_t)255;
  return r;
}

extern "C" void kernel_launch(void* const* d_in, const int* in_sizes, int n_in,
                              void* d_out, int out_size, void* d_ws, size_t ws_size,
                              hipStream_t stream) {
  (void)in_sizes; (void)n_in; (void)out_size; (void)ws_size;
  const float* x    = (const float*)d_in[0];
  const int*   ei   = (const int*)  d_in[1];
  const float* ea   = (const float*)d_in[2];
  const int*   rev  = (const int*)  d_in[3];
  const float* pka  = (const float*)d_in[4];
  const float* Wi1w = (const float*)d_in[5];
  const float* Wi1b = (const float*)d_in[6];
  const float* Wh1w = (const float*)d_in[7];
  const float* Wh1b = (const float*)d_in[8];
  const float* Wo1w = (const float*)d_in[9];
  const float* Wo1b = (const float*)d_in[10];
  const float* Wi3w = (const float*)d_in[11];
  const float* Wi3b = (const float*)d_in[12];
  const float* Wh3w = (const float*)d_in[13];
  const float* Wh3b = (const float*)d_in[14];
  const float* Wo3w = (const float*)d_in[15];
  const float* Wo3b = (const float*)d_in[16];
  const float* gw   = (const float*)d_in[17];
  const float* gb   = (const float*)d_in[18];
  const float* c1w  = (const float*)d_in[19];
  const float* c1b  = (const float*)d_in[20];
  const float* c2w  = (const float*)d_in[21];
  const float* c2b  = (const float*)d_in[22];
  const float* r1w  = (const float*)d_in[23];
  const float* r1b  = (const float*)d_in[24];
  const float* r2w  = (const float*)d_in[25];
  const float* r2b  = (const float*)d_in[26];

  const int* srcp = ei;
  const int* dstp = ei + N_EDGES;

  // ---- workspace carve ----
  char* wp = (char*)d_ws;
  __bf16* H0e  = (__bf16*)ws_carve(wp, (size_t)N_EDGES * HID * 2);
  __bf16* HeA  = (__bf16*)ws_carve(wp, (size_t)N_EDGES * HID * 2);
  __bf16* HeB  = (__bf16*)ws_carve(wp, (size_t)N_EDGES * HID * 2);
  float* accN  = (float*) ws_carve(wp, (size_t)N_NODES * HID * 4);
  float* hstat = (float*) ws_carve(wp, (size_t)N_NODES * HID * 4);
  float* hcur  = (float*) ws_carve(wp, (size_t)N_NODES * HID * 4);
  float* hupd  = (float*) ws_carve(wp, (size_t)N_NODES * HID * 4);
  float* tbuf  = (float*) ws_carve(wp, (size_t)N_NODES * HID * 4);
  __bf16* bWi1 = (__bf16*)ws_carve(wp, 128 * 160 * 2);
  __bf16* bWh1 = (__bf16*)ws_carve(wp, 128 * 128 * 2);
  __bf16* bWo1 = (__bf16*)ws_carve(wp, 128 * 256 * 2);
  __bf16* bWi3 = (__bf16*)ws_carve(wp, 128 * 160 * 2);
  __bf16* bWh3 = (__bf16*)ws_carve(wp, 128 * 128 * 2);
  __bf16* bWo3 = (__bf16*)ws_carve(wp, 128 * 256 * 2);
  __bf16* bC1  = (__bf16*)ws_carve(wp, 128 * 128 * 2);
  __bf16* bR1  = (__bf16*)ws_carve(wp, 128 * 128 * 2);
  float* hrow  = (float*) ws_carve(wp, 128 * 4);
  float* scal  = (float*) ws_carve(wp, 256);
  float* cla_acc = scal + 0;
  float* reg_acc = scal + 1;
  int*   cnt     = (int*)(scal + 2);
  float* vals    = scal + 3;          // 8 floats
  int*   poss    = (int*)(scal + 11); // 8 ints
  int*   idxs    = (int*)(scal + 19); // 3 ints

  // ---- weight conversion (f32 -> bf16, Wi padded 144 -> 160) ----
  auto cvt = [&](const float* w, __bf16* o, int nout, int k, int kpad) {
    const int tot = nout * kpad;
    k_cvt_w<<<(tot + 255) / 256, 256, 0, stream>>>(w, o, nout, k, kpad);
  };
  cvt(Wi1w, bWi1, 128, 144, 160);
  cvt(Wh1w, bWh1, 128, 128, 128);
  cvt(Wo1w, bWo1, 128, 256, 256);
  cvt(Wi3w, bWi3, 128, 144, 160);
  cvt(Wh3w, bWh3, 128, 128, 128);
  cvt(Wo3w, bWo3, 128, 256, 256);
  cvt(c1w,  bC1,  128, 128, 128);
  cvt(r1w,  bR1,  128, 128, 128);

  k_zero_f<<<1, 64, 0, stream>>>(scal, 64);
  k_findpos<<<(N_NODES + 255) / 256, 256, 0, stream>>>(pka, cnt, poss, vals);
  k_sort3<<<1, 1, 0, stream>>>(cnt, poss, vals, idxs);

  const int gE     = (N_EDGES + 31) / 32;
  const int gN     = (N_NODES + 31) / 32;
  const int gScat  = (int)(((long)N_EDGES * HID + 255) / 256);
  const int gNodeE = (N_NODES * HID + 255) / 256;

  // ---- one BondMessagePassing forward ----
  auto run_bmp = [&](const float* xb,
                     const __bf16* bWi, const float* Wib,
                     const __bf16* bWh, const float* Whb2,
                     const __bf16* bWo, const float* Wob,
                     float* out) {
    // H0 = Wi@concat(x[src],ea)+b ; H = relu(H0)
    k_gemm<1, 1, 160><<<gE, 256, 0, stream>>>(
        xb, ea, nullptr, nullptr, srcp, nullptr,
        bWi, Wib, nullptr, nullptr, H0e, HeA, N_EDGES);
    __bf16* hin = HeA; __bf16* hout = HeB;
    for (int it = 0; it < 3; ++it) {
      k_zero_f<<<gNodeE, 256, 0, stream>>>(accN, N_NODES * HID);
      k_scatter<<<gScat, 256, 0, stream>>>(hin, srcp, dstp, accN); // acc[dst]+=H[src]
      // H = relu(H0 + Wh@(acc[src]-H[rev]) + b)
      k_gemm<2, 2, 128><<<gE, 256, 0, stream>>>(
          nullptr, nullptr, accN, hin, srcp, rev,
          bWh, Whb2, H0e, nullptr, nullptr, hout, N_EDGES);
      __bf16* tsw = hin; hin = hout; hout = tsw;
    }
    // Magg = scatter(H by dst); Msel = iso ? x : Magg
    k_zero_f<<<gNodeE, 256, 0, stream>>>(accN, N_NODES * HID);
    k_scatter<<<gScat, 256, 0, stream>>>(hin, nullptr, dstp, accN); // acc[dst]+=H[e]
    k_msel<<<(N_NODES + 7) / 8, 256, 0, stream>>>(accN, xb);
    // out = relu(Wo@concat(x, Msel) + b)
    k_gemm<3, 0, 256><<<gN, 256, 0, stream>>>(
        xb, nullptr, accN, nullptr, nullptr, nullptr,
        bWo, Wob, nullptr, out, nullptr, nullptr, N_NODES);
  };

  run_bmp(x,     bWi1, Wi1b, bWh1, Wh1b, bWo1, Wo1b, hstat);
  run_bmp(hstat, bWi3, Wi3b, bWh3, Wh3b, bWo3, Wo3b, hcur);

  float* out_logits = (float*)d_out;                 // [N,2]
  float* out_pka    = (float*)d_out + 2 * N_NODES;   // [N]
  float* out_scal   = (float*)d_out + 3 * N_NODES;   // total, loss_cla, loss_reg
  const float ratio = ((float)N_NODES - 1.0f) / (1.0f + 1e-6f);

  for (int k = 0; k < 3; ++k) {
    // classifier: t = relu(c1@h+b); logits = c2@t+b  (last iter's logits persist)
    k_gemm<0, 0, 128><<<gN, 256, 0, stream>>>(
        hcur, nullptr, nullptr, nullptr, nullptr, nullptr,
        bC1, c1b, nullptr, tbuf, nullptr, nullptr, N_NODES);
    k_logits<<<(N_NODES + 255) / 256, 256, 0, stream>>>(tbuf, c2w, c2b, out_logits);
    k_cla<<<512, 256, 0, stream>>>(out_logits, idxs, k, ratio, cla_acc);
    // regressor(row idx) + huber; gate; gated row
    k_rowhead<<<1, 128, 0, stream>>>(hcur, idxs, k, pka,
                                     r1w, r1b, r2w, r2b, gw, gb, hrow, reg_acc);
    // h_upd = h_static ; h_upd[idx] = h_cur[idx]*g + h_cur[idx]
    k_copypatch<<<gNodeE, 256, 0, stream>>>(hstat, hrow, idxs, k, hupd);
    // h_cur = BMP(gcn3, h_upd)
    run_bmp(hupd, bWi3, Wi3b, bWh3, Wh3b, bWo3, Wo3b, hcur);
  }

  // final_pka = r2 @ relu(r1@h_cur + b) + b
  k_gemm<0, 0, 128><<<gN, 256, 0, stream>>>(
      hcur, nullptr, nullptr, nullptr, nullptr, nullptr,
      bR1, r1b, nullptr, tbuf, nullptr, nullptr, N_NODES);
  k_pka<<<(N_NODES + 255) / 256, 256, 0, stream>>>(tbuf, r2w, r2b, out_pka);
  k_final<<<1, 1, 0, stream>>>(cla_acc, reg_acc, out_scal);
}
